// Gemma4AudioModel_33646773797312
// MI455X (gfx1250) — compile-verified, matrix operand order checked by reading, not run on hardware
//
#include <hip/hip_runtime.h>
#include <hip/hip_bf16.h>

// ---------------------------------------------------------------------------
// Types for CDNA5 WMMA (wave32, 16x16x32 bf16 -> f32)
// ---------------------------------------------------------------------------
typedef __attribute__((ext_vector_type(16))) __bf16 vbf16;
typedef __attribute__((ext_vector_type(8)))  float  v8f;

__device__ __forceinline__ unsigned short f2bf(float f) {
    unsigned int u = __float_as_uint(f);
    u += 0x7FFFu + ((u >> 16) & 1u);          // round-to-nearest-even
    return (unsigned short)(u >> 16);
}
#if defined(__has_builtin) && __has_builtin(__builtin_amdgcn_cvt_pk_bf16_f32)
typedef __attribute__((ext_vector_type(2))) __bf16 v2bf;
__device__ __forceinline__ unsigned int pack2bf(float lo, float hi) {
    v2bf p = __builtin_amdgcn_cvt_pk_bf16_f32(lo, hi);   // v_cvt_pk_bf16_f32
    return __builtin_bit_cast(unsigned int, p);
}
#else
__device__ __forceinline__ unsigned int pack2bf(float lo, float hi) {
    return (unsigned int)f2bf(lo) | ((unsigned int)f2bf(hi) << 16);
}
#endif
__device__ __forceinline__ float bf2f(unsigned short s) {
    return __uint_as_float(((unsigned int)s) << 16);
}

// LDS tile row strides, padded to kill bank conflicts on fragment loads
// (64 banks x 4B; stride mod 64 dwords must spread the 16 row-lanes).
#define LDG  72    // GEMM tiles:      72 sh = 36 dw  -> 16 distinct groups
#define LDA  136   // attn 128-wide:  136 sh = 68 dw ≡ 4 -> bank 4*i, perfect
#define LDB  264   // attn 256-wide:  264 sh = 132 dw ≡ 4 -> perfect

// A/B fragment loader. Layout (contiguous k order, same for A and B): lane L holds row (base+L%16),
// 16 contiguous K values starting at kbase + (L/16)*16. For C = A @ B^T both
// fragments load identically from row-major LDS tiles.
__device__ __forceinline__ vbf16 ldfrag(const unsigned short* base, int row,
                                        int ld, int kbase, int lane) {
    const unsigned short* p =
        base + (row + (lane & 15)) * ld + kbase + ((lane >> 4) << 4);
    return *reinterpret_cast<const vbf16*>(p);
}

__device__ __forceinline__ v8f wmma_bf16(vbf16 a, vbf16 b, v8f c) {
    return __builtin_amdgcn_wmma_f32_16x16x32_bf16(
        false, a, false, b, (short)0, c, false, false);
}

// ---------------------------------------------------------------------------
// Generic tiled GEMM: C[M x N] = A[M x K] @ W[N x K]^T
//   A_BF16:     A is bf16-as-u16 (else f32); W always f32
//   SCALE_MODE: 0 none, 1 q-scale (per-dim softplus), 2 k-scale
//   OUT_BF16:   store bf16 (else f32)
// Block: 128x128 tile, 256 threads (8 waves), K-step 64 (2 WMMA substeps).
// ---------------------------------------------------------------------------
template <int A_BF16, int SCALE_MODE, int OUT_BF16>
__global__ __launch_bounds__(256) void gemm_wmma(
    const void* __restrict__ A, const float* __restrict__ W,
    void* __restrict__ Cout, int M, int N, int K,
    const float* __restrict__ pds)
{
    __shared__ unsigned short As[128 * LDG];   // 18 KB
    __shared__ unsigned short Ws[128 * LDG];   // 18 KB
    const int tid = threadIdx.x, lane = tid & 31, wave = tid >> 5;
    const int m0 = blockIdx.x * 128, n0 = blockIdx.y * 128;
    const float* Af = (const float*)A;
    const unsigned short* Ab = (const unsigned short*)A;

    const v8f zero8 = {0.f, 0.f, 0.f, 0.f, 0.f, 0.f, 0.f, 0.f};
    v8f acc[8];
#pragma unroll
    for (int i = 0; i < 8; i++) acc[i] = zero8;

    for (int k0 = 0; k0 < K; k0 += 64) {
        // ---- stage W (f32 -> bf16, float4 loads, b64 LDS stores) ----
#pragma unroll
        for (int i = 0; i < 8; i++) {         // 2048 float4 chunks
            int v = tid + i * 256;
            int r = v >> 4, c = (v & 15) << 2;
            float4 w = *(const float4*)(W + (size_t)(n0 + r) * K + k0 + c);
            uint2 wp;
            wp.x = pack2bf(w.x, w.y);
            wp.y = pack2bf(w.z, w.w);
            *(uint2*)(Ws + r * LDG + c) = wp;
        }
        // ---- stage A ----
        if constexpr (A_BF16) {
#pragma unroll
            for (int i = 0; i < 4; i++) {     // 1024 uint4 (8-elem) chunks
                int v = tid + i * 256;
                int r = v >> 3, c = (v & 7) << 3;
                uint4 av = {0, 0, 0, 0};
                if (m0 + r < M)
                    av = *(const uint4*)(Ab + (size_t)(m0 + r) * K + k0 + c);
                *(uint4*)(As + r * LDG + c) = av;
            }
        } else {
#pragma unroll
            for (int i = 0; i < 8; i++) {
                int v = tid + i * 256;
                int r = v >> 4, c = (v & 15) << 2;
                uint2 ap = {0, 0};
                if (m0 + r < M) {
                    float4 f = *(const float4*)(Af + (size_t)(m0 + r) * K + k0 + c);
                    ap.x = pack2bf(f.x, f.y);
                    ap.y = pack2bf(f.z, f.w);
                }
                *(uint2*)(As + r * LDG + c) = ap;
            }
        }
        // prefetch next K tile while WMMAs run (global_prefetch_b8)
        if (k0 + 64 < K) {
            int r = tid >> 1, c = (tid & 1) << 5;
            __builtin_prefetch(W + (size_t)(n0 + r) * K + k0 + 64 + c, 0, 0);
            if (m0 + r < M) {
                if constexpr (A_BF16)
                    __builtin_prefetch(Ab + (size_t)(m0 + r) * K + k0 + 64 + c, 0, 0);
                else
                    __builtin_prefetch(Af + (size_t)(m0 + r) * K + k0 + 64 + c, 0, 0);
            }
        }
        __syncthreads();
#pragma unroll
        for (int kb = 0; kb < 64; kb += 32) {
            vbf16 a = ldfrag(As, wave * 16, LDG, kb, lane);
#pragma unroll
            for (int nt = 0; nt < 8; nt++) {
                vbf16 b = ldfrag(Ws, nt * 16, LDG, kb, lane);
                acc[nt] = wmma_bf16(a, b, acc[nt]);
            }
        }
        __syncthreads();
    }

    // Epilogue. C layout: lane 0-15 -> rows r, N=lane; lane 16-31 -> rows r+8.
    const int half = lane >> 4, lr = lane & 15;
    const float QSCALE = 0.12751774f;            // D^-0.5 / ln 2
    const float KSCALE = 1.8946372f;             // log1p(e) / ln 2
#pragma unroll
    for (int nt = 0; nt < 8; nt++) {
#pragma unroll
        for (int r = 0; r < 8; r++) {
            int gm = m0 + wave * 16 + r + 8 * half;
            int gn = n0 + nt * 16 + lr;
            if (gm < M) {
                float v = acc[nt][r];
                if constexpr (SCALE_MODE == 1) {
                    float sp = logf(1.0f + __expf(pds[gn & 127]));
                    v *= QSCALE * sp;
                } else if constexpr (SCALE_MODE == 2) {
                    v *= KSCALE;
                }
                if constexpr (OUT_BF16)
                    ((unsigned short*)Cout)[(size_t)gm * N + gn] = f2bf(v);
                else
                    ((float*)Cout)[(size_t)gm * N + gn] = v;
            }
        }
    }
}

// ---------------------------------------------------------------------------
// Fused chunked relative attention. One workgroup per (b, h, block).
// 256 threads = 8 waves; wave w owns query rows [w*16, w*16+16).
// Context padded to 256 cols (col 255 forced to -inf pre-softmax).
// rel-shift: shifted[q,k] == bd[(j>>8)][(j&255)] with j = q*255+k.
// ---------------------------------------------------------------------------
__global__ __launch_bounds__(256) void attn_kernel(
    const unsigned short* __restrict__ qb,   // [16384][1536] bf16
    const unsigned short* __restrict__ kb,
    const unsigned short* __restrict__ vb,
    const unsigned short* __restrict__ relk, // [255][1536] bf16
    unsigned short* __restrict__ outb)       // [16384][1536] bf16
{
    extern __shared__ unsigned short smem[];
    unsigned short* Qs = smem;                 // 128 rows x LDA
    unsigned short* Ks = Qs + 128 * LDA;       // 256 rows x LDA (ctx rows)
    unsigned short* Rs = Ks + 256 * LDA;       // 256 x LDA rel -> later VT 128 x LDB
    unsigned short* Bs = Rs + 256 * LDA;       // 128 x LDB bd  -> later P
    const int tid = threadIdx.x, lane = tid & 31, wave = tid >> 5;
    const int half = lane >> 4, lr = lane & 15;
    const int bid = blockIdx.x;
    const int b = bid / (12 * 32);
    const int h = (bid % (12 * 32)) / 32;
    const int n = bid % 32;
    const size_t rowbase = (size_t)b * 4096;
    const int hoff = h * 128;

    // --- stage Q, K-context, Rel (vectorized uint4 = 8 bf16) ---
    for (int c = tid; c < 128 * 16; c += 256) {
        int r = c >> 4, cc = (c & 15) << 3;
        *(uint4*)(Qs + r * LDA + cc) =
            *(const uint4*)(qb + (rowbase + n * 128 + r) * 1536 + hoff + cc);
    }
    for (int c = tid; c < 256 * 16; c += 256) {
        int r = c >> 4, cc = (c & 15) << 3;
        int src = n * 128 + r - 127;            // PAST = 127
        uint4 val = {0, 0, 0, 0};
        if (r < 255 && src >= 0 && src < 4096)
            val = *(const uint4*)(kb + (rowbase + src) * 1536 + hoff + cc);
        *(uint4*)(Ks + r * LDA + cc) = val;
    }
    for (int c = tid; c < 256 * 16; c += 256) {
        int r = c >> 4, cc = (c & 15) << 3;
        uint4 val = {0, 0, 0, 0};
        if (r < 255)
            val = *(const uint4*)(relk + (size_t)r * 1536 + hoff + cc);
        *(uint4*)(Rs + r * LDA + cc) = val;
    }
    __syncthreads();

    const v8f zero8 = {0.f, 0.f, 0.f, 0.f, 0.f, 0.f, 0.f, 0.f};
    v8f acc[16];

    // --- phase 1: bd = Q @ Rel^T  (128 x 256) -> Bs (bf16) ---
#pragma unroll
    for (int i = 0; i < 16; i++) acc[i] = zero8;
#pragma unroll
    for (int kt = 0; kt < 4; kt++) {
        vbf16 a = ldfrag(Qs, wave * 16, LDA, kt * 32, lane);
#pragma unroll
        for (int nt = 0; nt < 16; nt++) {
            vbf16 bf = ldfrag(Rs, nt * 16, LDA, kt * 32, lane);
            acc[nt] = wmma_bf16(a, bf, acc[nt]);
        }
    }
#pragma unroll
    for (int nt = 0; nt < 16; nt++)
#pragma unroll
        for (int r = 0; r < 8; r++) {
            int qq = wave * 16 + r + 8 * half;
            Bs[qq * LDB + nt * 16 + lr] = f2bf(acc[nt][r]);
        }
    __syncthreads();

    // --- stage V transposed into Rs: VT[d][ctx] (Rel no longer needed) ---
    for (int c = tid; c < 256 * 16; c += 256) {
        int r = c >> 4, cc = (c & 15) << 3;     // ctx row r, d-base cc
        int src = n * 128 + r - 127;
        uint4 tv = {0, 0, 0, 0};
        if (r < 255 && src >= 0 && src < 4096)
            tv = *(const uint4*)(vb + (rowbase + src) * 1536 + hoff + cc);
        unsigned int w4[4] = {tv.x, tv.y, tv.z, tv.w};
#pragma unroll
        for (int i = 0; i < 4; i++) {
            Rs[(cc + 2 * i) * LDB + r]     = (unsigned short)(w4[i] & 0xFFFF);
            Rs[(cc + 2 * i + 1) * LDB + r] = (unsigned short)(w4[i] >> 16);
        }
    }

    // --- phase 2: ac = Q @ K^T, add shifted bd, softcap ---
#pragma unroll
    for (int i = 0; i < 16; i++) acc[i] = zero8;
#pragma unroll
    for (int kt = 0; kt < 4; kt++) {
        vbf16 a = ldfrag(Qs, wave * 16, LDA, kt * 32, lane);
#pragma unroll
        for (int nt = 0; nt < 16; nt++) {
            vbf16 bf = ldfrag(Ks, nt * 16, LDA, kt * 32, lane);
            acc[nt] = wmma_bf16(a, bf, acc[nt]);
        }
    }
#pragma unroll
    for (int nt = 0; nt < 16; nt++)
#pragma unroll
        for (int r = 0; r < 8; r++) {
            int qq = wave * 16 + r + 8 * half;
            int kk = nt * 16 + lr;
            float l = acc[nt][r];
            if (kk < 255) {
                int j = qq * 255 + kk;          // rel-shift linear index
                l += bf2f(Bs[(j >> 8) * LDB + (j & 255)]);
                l = tanhf(l * 0.02f) * 50.0f;   // softcap 50
            } else {
                l = -3.0e38f;                   // padded column
            }
            acc[nt][r] = l;
        }
    __syncthreads();   // all bd reads done; VT staged

    // --- softmax over 256 cols per query row (16-lane shuffle groups) ---
    float srow[8];
#pragma unroll
    for (int r = 0; r < 8; r++) {
        float m = -3.0e38f;
#pragma unroll
        for (int nt = 0; nt < 16; nt++) m = fmaxf(m, acc[nt][r]);
#pragma unroll
        for (int d = 8; d >= 1; d >>= 1) m = fmaxf(m, __shfl_xor(m, d, 32));
        float s = 0.f;
#pragma unroll
        for (int nt = 0; nt < 16; nt++) {
            float p = __expf(acc[nt][r] - m);
            acc[nt][r] = p;
            s += p;
        }
#pragma unroll
        for (int d = 8; d >= 1; d >>= 1) s += __shfl_xor(s, d, 32);
        srow[r] = 1.0f / s;
    }

    // --- write P (bf16) into Bs ---
#pragma unroll
    for (int nt = 0; nt < 16; nt++)
#pragma unroll
        for (int r = 0; r < 8; r++) {
            int qq = wave * 16 + r + 8 * half;
            Bs[qq * LDB + nt * 16 + lr] = f2bf(acc[nt][r] * srow[r]);
        }
    __syncthreads();

    // --- out = P[128x256] @ V[256x128] (VT rows are d, contiguous in ctx) ---
    v8f oacc[8];
#pragma unroll
    for (int i = 0; i < 8; i++) oacc[i] = zero8;
#pragma unroll
    for (int kt = 0; kt < 8; kt++) {
        vbf16 a = ldfrag(Bs, wave * 16, LDB, kt * 32, lane);
#pragma unroll
        for (int nt = 0; nt < 8; nt++) {
            vbf16 bf = ldfrag(Rs, nt * 16, LDB, kt * 32, lane);
            oacc[nt] = wmma_bf16(a, bf, oacc[nt]);
        }
    }
#pragma unroll
    for (int nt = 0; nt < 8; nt++)
#pragma unroll
        for (int r = 0; r < 8; r++) {
            int qq = wave * 16 + r + 8 * half;
            int dd = nt * 16 + lr;
            outb[(rowbase + n * 128 + qq) * 1536 + hoff + dd] = f2bf(oacc[nt][r]);
        }
}

// ---------------------------------------------------------------------------
// Host launcher
// ---------------------------------------------------------------------------
extern "C" void kernel_launch(void* const* d_in, const int* in_sizes, int n_in,
                              void* d_out, int out_size, void* d_ws, size_t ws_size,
                              hipStream_t stream) {
    (void)in_sizes; (void)n_in; (void)out_size; (void)ws_size;
    const float* hs    = (const float*)d_in[0];
    const float* pe    = (const float*)d_in[1];
    const float* Wq    = (const float*)d_in[2];
    const float* Wk    = (const float*)d_in[3];
    const float* Wv    = (const float*)d_in[4];
    const float* Wrel  = (const float*)d_in[5];
    const float* pds   = (const float*)d_in[6];
    const float* Wpost = (const float*)d_in[7];

    const int M = 4 * 4096, N = 1536, K = 1536;
    unsigned short* qbuf = (unsigned short*)d_ws;
    unsigned short* kbuf = qbuf + (size_t)M * N;
    unsigned short* vbuf = kbuf + (size_t)M * N;
    unsigned short* abuf = vbuf + (size_t)M * N;
    unsigned short* rbuf = abuf + (size_t)M * N;

    dim3 blk(256);
    dim3 g(M / 128, N / 128);
    gemm_wmma<0, 1, 1><<<g, blk, 0, stream>>>(hs, Wq, qbuf, M, N, K, pds);
    gemm_wmma<0, 2, 1><<<g, blk, 0, stream>>>(hs, Wk, kbuf, M, N, K, nullptr);
    gemm_wmma<0, 0, 1><<<g, blk, 0, stream>>>(hs, Wv, vbuf, M, N, K, nullptr);
    dim3 gr(2, N / 128);
    gemm_wmma<0, 0, 1><<<gr, blk, 0, stream>>>(pe, Wrel, rbuf, 255, N, K, nullptr);

    // LDS: (128 + 256 + 256)*LDA + 128*LDB shorts = 241664 B (236 KB <= 320 KB/WGP)
    const int smem = (128 * LDA + 256 * LDA + 256 * LDA + 128 * LDB) * 2;
    hipFuncSetAttribute(reinterpret_cast<const void*>(attn_kernel),
                        hipFuncAttributeMaxDynamicSharedMemorySize, smem);
    attn_kernel<<<dim3(4 * 12 * 32), blk, smem, stream>>>(qbuf, kbuf, vbuf, rbuf, abuf);

    gemm_wmma<1, 0, 0><<<g, blk, 0, stream>>>(abuf, Wpost, (float*)d_out, M, N, K, nullptr);
}
